// TransformerEncoderLayer_34626026341033
// MI455X (gfx1250) — matrix operand order repaired and checked by reading, not compiled, not run on hardware
//
#include <hip/hip_runtime.h>
#include <hip/hip_bf16.h>

typedef __attribute__((ext_vector_type(16))) _Float16 v16h;
typedef __attribute__((ext_vector_type(8)))  _Float16 v8h;
typedef __attribute__((ext_vector_type(8)))  float    v8f;

// ---------------------------------------------------------------------------
// Prep: q = half(src + pos), src_h = half(src)
// ---------------------------------------------------------------------------
__global__ __launch_bounds__(256) void prep_inputs(
    const float* __restrict__ src, const float* __restrict__ pos,
    _Float16* __restrict__ q_h, _Float16* __restrict__ src_h, int total)
{
    int i = blockIdx.x * blockDim.x + threadIdx.x;
    if (i < total) {
        float s = src[i];
        src_h[i] = (_Float16)s;
        q_h[i]   = (_Float16)(s + pos[i]);
    }
}

// ---------------------------------------------------------------------------
// Weight convert+transpose: W[K][N] f32 -> WT[N][K] f16
// ---------------------------------------------------------------------------
__global__ __launch_bounds__(256) void transpose_to_half(
    const float* __restrict__ W, _Float16* __restrict__ WT, int K, int N)
{
    int i = blockIdx.x * blockDim.x + threadIdx.x;
    if (i < K * N) {
        int k = i / N, n = i - k * N;
        WT[(size_t)n * K + k] = (_Float16)W[i];
    }
}

// ---------------------------------------------------------------------------
// WMMA GEMM: C[M x N] = A_h[M x K] * BT_h[N x K]^T + bias
// One 16x64 C tile per wave32: a single A fragment is reused across 4
// accumulators / 4 B column fragments (4 v_wmma per A load).
// A fragment per ISA layout: lanes 0-15 hold K = k+{0..7, 16..23},
// lanes 16-31 hold K = k+{8..15, 24..31}. B fragment: same K split as A
// (column = lane&15). Two 16B loads each.
// ---------------------------------------------------------------------------
template<bool RELU_HALF>
__global__ __launch_bounds__(256) void gemm_wmma_f16(
    const _Float16* __restrict__ A, const _Float16* __restrict__ BT,
    const float* __restrict__ bias,
    float* __restrict__ Cf, _Float16* __restrict__ Ch,
    int M, int N, int K)
{
    const int lane = threadIdx.x & 31;
    const int wave = blockIdx.x * (blockDim.x >> 5) + (threadIdx.x >> 5);
    const int tilesN = N >> 6;                     // 64-wide N tiles
    const int tilesM = (M + 15) >> 4;
    if (wave >= tilesM * tilesN) return;           // wave-uniform exit
    const int tm = (wave / tilesN) << 4;
    const int tn = (wave % tilesN) << 6;

    int arow = tm + (lane & 15);
    if (arow > M - 1) arow = M - 1;                // clamp tail rows (loads only)
    const int hsel = (lane >> 4) & 1;
    const _Float16* aptr = A  + (size_t)arow * K + (hsel ? 8  : 0);
    const int bcol = tn + (lane & 15);
    const _Float16* bptr = BT + (size_t)bcol * K + (hsel ? 8  : 0);
    const size_t bstride = (size_t)16 * K;         // 16 columns of BT

    v8f c0 = {}, c1 = {}, c2 = {}, c3 = {};
    for (int k = 0; k < K; k += 32) {
        __builtin_prefetch(aptr + k + 32, 0, 0);   // global_prefetch_b8
        // A fragment: loaded once, reused by 4 WMMAs
        v8h alo = *(const v8h*)(aptr + k);
        v8h ahi = *(const v8h*)(aptr + k + 16);
        v16h a;
#pragma unroll
        for (int i = 0; i < 8; ++i) { a[i] = alo[i]; a[i + 8] = ahi[i]; }

        // 4 B column fragments -> 4 accumulators
        v16h b[4];
#pragma unroll
        for (int j = 0; j < 4; ++j) {
            const _Float16* bj = bptr + (size_t)j * bstride + k;
            __builtin_prefetch(bj + 32, 0, 0);
            v8h b0 = *(const v8h*)(bj);
            v8h b1 = *(const v8h*)(bj + 16);
#pragma unroll
            for (int i = 0; i < 8; ++i) { b[j][i] = b0[i]; b[j][i + 8] = b1[i]; }
        }
        c0 = __builtin_amdgcn_wmma_f32_16x16x32_f16(false, a, false, b[0], (short)0, c0, false, false);
        c1 = __builtin_amdgcn_wmma_f32_16x16x32_f16(false, a, false, b[1], (short)0, c1, false, false);
        c2 = __builtin_amdgcn_wmma_f32_16x16x32_f16(false, a, false, b[2], (short)0, c2, false, false);
        c3 = __builtin_amdgcn_wmma_f32_16x16x32_f16(false, a, false, b[3], (short)0, c3, false, false);
    }

    const int rbase = tm + (hsel ? 8 : 0);
    v8f acc[4] = {c0, c1, c2, c3};
#pragma unroll
    for (int j = 0; j < 4; ++j) {
        const int col = tn + j * 16 + (lane & 15);
        const float bc = bias ? bias[col] : 0.f;
#pragma unroll
        for (int i = 0; i < 8; ++i) {
            int row = rbase + i;
            if (row < M) {
                float v = acc[j][i] + bc;
                if (RELU_HALF) {
                    v = v > 0.f ? v : 0.f;
                    Ch[(size_t)row * N + col] = (_Float16)v;
                } else {
                    Cf[(size_t)row * N + col] = v;
                }
            }
        }
    }
}

// ---------------------------------------------------------------------------
// Value reshape + padding mask: vproj[B*LQ][256] -> v_r[(b*8+h)][LQ][32]
// ---------------------------------------------------------------------------
__global__ __launch_bounds__(256) void reshape_value(
    const float* __restrict__ vproj, const unsigned char* __restrict__ mask,
    float* __restrict__ v_r, int LQ)
{
    int i = blockIdx.x * blockDim.x + threadIdx.x;
    int total = 2 * LQ * 256;
    if (i < total) {
        int c = i & 255;
        int m = i >> 8;                  // b*LQ + l
        int b = m / LQ, l = m - b * LQ;
        int h = c >> 5, d = c & 31;
        float v = vproj[i] * (mask[(size_t)b * LQ + l] ? 1.f : 0.f);
        v_r[(((size_t)(b * 8 + h)) * LQ + l) * 32 + d] = v;
    }
}

// ---------------------------------------------------------------------------
// Deformable attention core: one wave per (b,q,h); lanes = HEAD_DIM channels.
// Softmax over 16 (level,point) weights, 16 bilinear samples, weighted sum.
// ---------------------------------------------------------------------------
__global__ __launch_bounds__(256) void deform_sample_kernel(
    const float* __restrict__ v_r,      // [2*8][LQ][32]
    const float* __restrict__ offp,     // [M][256]
    const float* __restrict__ attnlog,  // [M][128]
    const float* __restrict__ refp,     // [2][LQ][4][2]
    _Float16* __restrict__ out_h,       // [M][256]
    int LQ)
{
    const int lane = threadIdx.x & 31;
    const int wv = (blockIdx.x * blockDim.x + threadIdx.x) >> 5;
    if (wv >= 2 * LQ * 8) return;
    const int h = wv & 7;
    const int t = wv >> 3;              // b*LQ + q
    const int b = t / LQ;

    // softmax over the 16 logits (lane-redundant, broadcast loads)
    const float* al = attnlog + (size_t)t * 128 + h * 16;
    float w[16], mx = -1e30f;
#pragma unroll
    for (int i = 0; i < 16; ++i) { w[i] = al[i]; mx = fmaxf(mx, w[i]); }
    float s = 0.f;
#pragma unroll
    for (int i = 0; i < 16; ++i) { w[i] = __expf(w[i] - mx); s += w[i]; }
    const float inv = 1.f / s;

    constexpr int LH[4]     = {100, 50, 25, 13};
    constexpr int LSTART[4] = {0, 10000, 12500, 13125};

    const float* ofs   = offp + (size_t)t * 256 + h * 32;
    const float* rp    = refp + (size_t)t * 8;
    const float* vbase = v_r + ((size_t)(b * 8 + h) * LQ) * 32;

    float acc = 0.f;
#pragma unroll
    for (int l = 0; l < 4; ++l) {
        const int   Hd = LH[l], Wd = LH[l];
        const float rx = rp[l * 2 + 0], ry = rp[l * 2 + 1];
        const float invD = 1.f / (float)Wd;      // square levels: W == H
        const float* vl = vbase + (size_t)LSTART[l] * 32;
#pragma unroll
        for (int p = 0; p < 4; ++p) {
            float lx = rx + ofs[(l * 4 + p) * 2 + 0] * invD;
            float ly = ry + ofs[(l * 4 + p) * 2 + 1] * invD;
            float x = lx * Wd - 0.5f, y = ly * Hd - 0.5f;
            float x0f = floorf(x), y0f = floorf(y);
            float wx = x - x0f, wy = y - y0f;
            int x0 = (int)x0f, y0 = (int)y0f;
            bool xi0 = (x0 >= 0) & (x0 < Wd);
            bool xi1 = (x0 + 1 >= 0) & (x0 + 1 < Wd);
            bool yi0 = (y0 >= 0) & (y0 < Hd);
            bool yi1 = (y0 + 1 >= 0) & (y0 + 1 < Hd);
            float v00 = 0.f, v01 = 0.f, v10 = 0.f, v11 = 0.f;
            if (xi0 & yi0) v00 = vl[((size_t)y0 * Wd + x0) * 32 + lane];
            if (xi1 & yi0) v01 = vl[((size_t)y0 * Wd + x0 + 1) * 32 + lane];
            if (xi0 & yi1) v10 = vl[((size_t)(y0 + 1) * Wd + x0) * 32 + lane];
            if (xi1 & yi1) v11 = vl[((size_t)(y0 + 1) * Wd + x0 + 1) * 32 + lane];
            acc += (w[l * 4 + p] * inv) *
                   (v00 * (1.f - wx) * (1.f - wy) + v01 * wx * (1.f - wy) +
                    v10 * (1.f - wx) * wy         + v11 * wx * wy);
        }
    }
    out_h[(size_t)t * 256 + h * 32 + lane] = (_Float16)acc;
}

// ---------------------------------------------------------------------------
// out = LayerNorm(x + y) * g + be ; optional f16 copy for next GEMM's A.
// One block (256 threads) per row of 256 channels.
// ---------------------------------------------------------------------------
__global__ __launch_bounds__(256) void add_layernorm(
    const float* __restrict__ x, const float* __restrict__ y,
    const float* __restrict__ g, const float* __restrict__ be,
    float* __restrict__ outf, _Float16* __restrict__ outh)
{
    const int row = blockIdx.x, c = threadIdx.x;
    const size_t idx = (size_t)row * 256 + c;
    float v = x[idx] + y[idx];
    __shared__ float red[256];
    red[c] = v; __syncthreads();
    for (int s = 128; s > 0; s >>= 1) { if (c < s) red[c] += red[c + s]; __syncthreads(); }
    float mu = red[0] * (1.f / 256.f);
    __syncthreads();
    float d = v - mu;
    red[c] = d * d; __syncthreads();
    for (int s = 128; s > 0; s >>= 1) { if (c < s) red[c] += red[c + s]; __syncthreads(); }
    float var = red[0] * (1.f / 256.f);
    float o = d * rsqrtf(var + 1e-5f) * g[c] + be[c];
    outf[idx] = o;
    if (outh) outh[idx] = (_Float16)o;
}

// ---------------------------------------------------------------------------
extern "C" void kernel_launch(void* const* d_in, const int* in_sizes, int n_in,
                              void* d_out, int out_size, void* d_ws, size_t ws_size,
                              hipStream_t stream)
{
    const float* src  = (const float*)d_in[0];
    const float* pos  = (const float*)d_in[1];
    const float* refp = (const float*)d_in[2];
    /* d_in[3] spatial_shapes: compile-time constants */
    const unsigned char* mask = (const unsigned char*)d_in[4];
    const float* Wv   = (const float*)d_in[5];  const float* bv   = (const float*)d_in[6];
    const float* Woff = (const float*)d_in[7];  const float* boff = (const float*)d_in[8];
    const float* Wa   = (const float*)d_in[9];  const float* ba   = (const float*)d_in[10];
    const float* Wout = (const float*)d_in[11]; const float* bout = (const float*)d_in[12];
    const float* W1   = (const float*)d_in[13]; const float* b1   = (const float*)d_in[14];
    const float* W2   = (const float*)d_in[15]; const float* b2   = (const float*)d_in[16];
    const float* g1   = (const float*)d_in[17]; const float* be1  = (const float*)d_in[18];
    const float* g2   = (const float*)d_in[19]; const float* be2  = (const float*)d_in[20];
    (void)n_in; (void)out_size; (void)ws_size;

    const int M  = in_sizes[0] / 256;   // BS*LQ = 26588
    const int LQ = M / 2;
    const int elems = M * 256;

    char* ws = (char*)d_ws;
    size_t off = 0;
    auto carve = [&](size_t bytes) -> char* {
        char* p = ws + off; off = (off + bytes + 255) & ~(size_t)255; return p;
    };
    _Float16* q_h   = (_Float16*)carve((size_t)M * 256 * 2);  // reused as out_h
    _Float16* src_h = (_Float16*)carve((size_t)M * 256 * 2);  // reused as x1_h
    _Float16* WvT   = (_Float16*)carve(256 * 256 * 2);
    _Float16* WoffT = (_Float16*)carve(256 * 256 * 2);
    _Float16* WaT   = (_Float16*)carve(128 * 256 * 2);
    _Float16* WoutT = (_Float16*)carve(256 * 256 * 2);
    _Float16* W1T   = (_Float16*)carve(1024 * 256 * 2);
    _Float16* W2T   = (_Float16*)carve(256 * 1024 * 2);
    float* bufA     = (float*)carve((size_t)M * 256 * 4);     // vproj -> attn_proj
    float* v_r      = (float*)carve((size_t)M * 256 * 4);
    float* offOut   = (float*)carve((size_t)M * 256 * 4);     // offsets -> ffn out
    float* attnlog  = (float*)carve((size_t)M * 128 * 4);
    float* x1       = (float*)carve((size_t)M * 256 * 4);
    _Float16* h1_h  = (_Float16*)carve((size_t)M * 1024 * 2);

    prep_inputs<<<(elems + 255) / 256, 256, 0, stream>>>(src, pos, q_h, src_h, elems);
    transpose_to_half<<<(256 * 256 + 255) / 256, 256, 0, stream>>>(Wv,   WvT,   256, 256);
    transpose_to_half<<<(256 * 256 + 255) / 256, 256, 0, stream>>>(Woff, WoffT, 256, 256);
    transpose_to_half<<<(256 * 128 + 255) / 256, 256, 0, stream>>>(Wa,   WaT,   256, 128);
    transpose_to_half<<<(256 * 256 + 255) / 256, 256, 0, stream>>>(Wout, WoutT, 256, 256);
    transpose_to_half<<<(256 * 1024 + 255) / 256, 256, 0, stream>>>(W1,  W1T,  256, 1024);
    transpose_to_half<<<(1024 * 256 + 255) / 256, 256, 0, stream>>>(W2,  W2T,  1024, 256);

    auto gemm_blocks = [&](int N) {
        int waves = ((M + 15) / 16) * (N / 64);   // 16x64 tile per wave
        return (waves + 7) / 8;                   // 8 waves per 256-thread block
    };

    // value projection, reshape+mask into per-head layout
    gemm_wmma_f16<false><<<gemm_blocks(256), 256, 0, stream>>>(
        src_h, WvT, bv, bufA, nullptr, M, 256, 256);
    reshape_value<<<(elems + 255) / 256, 256, 0, stream>>>(bufA, mask, v_r, LQ);

    // sampling offsets and attention logits
    gemm_wmma_f16<false><<<gemm_blocks(256), 256, 0, stream>>>(
        q_h, WoffT, boff, offOut, nullptr, M, 256, 256);
    gemm_wmma_f16<false><<<gemm_blocks(128), 256, 0, stream>>>(
        q_h, WaT, ba, attnlog, nullptr, M, 128, 256);

    // deformable sampling (q_h no longer needed -> reuse as f16 head output)
    _Float16* out_h = q_h;
    {
        int waves  = 2 * LQ * 8;
        int blocks = (waves * 32 + 255) / 256;
        deform_sample_kernel<<<blocks, 256, 0, stream>>>(
            v_r, offOut, attnlog, refp, out_h, LQ);
    }

    // output projection (bufA reused), residual + LN1
    gemm_wmma_f16<false><<<gemm_blocks(256), 256, 0, stream>>>(
        out_h, WoutT, bout, bufA, nullptr, M, 256, 256);
    _Float16* x1_h = src_h;
    add_layernorm<<<M, 256, 0, stream>>>(src, bufA, g1, be1, x1, x1_h);

    // FFN: relu(x1 @ W1 + b1) @ W2 + b2, residual + LN2 -> d_out
    gemm_wmma_f16<true><<<gemm_blocks(1024), 256, 0, stream>>>(
        x1_h, W1T, b1, nullptr, h1_h, M, 1024, 256);
    float* ffn = offOut;
    gemm_wmma_f16<false><<<gemm_blocks(256), 256, 0, stream>>>(
        h1_h, W2T, b2, ffn, nullptr, M, 256, 1024);
    add_layernorm<<<M, 256, 0, stream>>>(x1, ffn, g2, be2, (float*)d_out, nullptr);
}
